// CustomLSTMClassifier_53919019434479
// MI455X (gfx1250) — matrix operand order repaired and checked by reading, not compiled, not run on hardware
//
#include <hip/hip_runtime.h>
#include <cstdint>
#include <cstddef>

// Problem constants
#define IN_SZ   64
#define HID     128
#define GATES   512      // 4*HID
#define BATCH   512
#define SEQ     1024
#define NCLS    4

#define BT      16       // batch rows per workgroup (= WMMA M)
#define WAVES   16       // waves per workgroup
#define TPB     (WAVES * 32)

// Padded LDS row strides (in halves) to kill bank conflicts:
// 136 halves = 272 B = 68 dwords -> fragment lane L starts at bank 4L (all distinct)
#define WROW    136      // Wh1/Wh2 row stride (128 data + 8 pad)
#define HROW    136      // h-state row stride
#define XROW    72       // x-stage row stride (64 data + 8 pad), 36 dwords
#define XBUFH   (BT * XROW)   // halves per x buffer (1152)

typedef __attribute__((ext_vector_type(16))) _Float16 v16h;
typedef __attribute__((ext_vector_type(8)))  _Float16 v8h;
typedef __attribute__((ext_vector_type(2)))  _Float16 v2h;
typedef __attribute__((ext_vector_type(8)))  float    v8f;

// ---- workspace layout (bytes) ----
#define WS_WX1H   0              // 512*64  f16 = 65536
#define WS_WH1H   65536          // 512*128 f16 = 131072
#define WS_WX2H   196608
#define WS_WH2H   327680
#define WS_BIAS1  458752         // 512 f32
#define WS_BIAS2  460800         // 512 f32

// ---- LDS layout (bytes) ----
#define LDS_WH1   0                          // 512x136 f16 = 139264
#define LDS_WH2   139264                     // 139264
#define LDS_GATES 278528                     // 16x512 f32 = 32768
#define LDS_H1    (278528 + 32768)           // 16x136 f16 = 4352
#define LDS_H2    (LDS_H1 + 4352)
#define LDS_X0    (LDS_H2 + 4352)            // 2 x (16x72 f16 = 2304)
#define LDS_TOTAL (LDS_X0 + 2 * 2304)        // 324608 B < 320 KB (327680)

// ---------------------------------------------------------------------------
// WMMA fragment loaders (16-bit, 16x16x32, wave32; contiguous k = 16*(lane/16) + s in both A and B)
// ---------------------------------------------------------------------------

// A (16xK tile, f16, row-major, row stride `rs` halves), k-chunk base k0.
__device__ __forceinline__ v16h load_afrag(const _Float16* base, int rs, int k0, int lane) {
    int row = lane & 15;
    int kA  = k0 + ((lane & 16) ? 16 : 0);
    v8h lo = *(const v8h*)(base + row * rs + kA);
    v8h hi = *(const v8h*)(base + row * rs + kA + 8);
    return __builtin_shufflevector(lo, hi, 0,1,2,3,4,5,6,7,8,9,10,11,12,13,14,15);
}

// B (32x16 tile) from W stored n-major with row stride `rs`: B[k][n] = W[n0+n, k0+k].
__device__ __forceinline__ v16h load_bfrag(const _Float16* w, int rs, int n0, int k0, int lane) {
    int n  = n0 + (lane & 15);
    int kb = k0 + ((lane & 16) ? 16 : 0);
    return *(const v16h*)(w + n * rs + kb);
}

__device__ __forceinline__ v8f wmma_f16(v16h a, v16h b, v8f c) {
    return __builtin_amdgcn_wmma_f32_16x16x32_f16(false, a, false, b, (short)0, c, false, false);
}

__device__ __forceinline__ float fast_rcp(float v) { return __builtin_amdgcn_rcpf(v); }
__device__ __forceinline__ float fsig(float v)  { return fast_rcp(1.0f + __expf(-v)); }
__device__ __forceinline__ float ftanh(float v) { float e = __expf(2.0f * v); return 1.0f - 2.0f * fast_rcp(e + 1.0f); }

// Async global->LDS copy (CDNA5 GLOBAL_LOAD_ASYNC_TO_LDS_B128, ASYNCcnt-tracked).
__device__ __forceinline__ void async_g2l_b128(uint32_t lds_byte, uint64_t gaddr) {
    asm volatile("global_load_async_to_lds_b128 %0, %1, off"
                 :: "v"(lds_byte), "v"(gaddr) : "memory");
}
__device__ __forceinline__ void wait_async0() {
    asm volatile("s_wait_asynccnt 0" ::: "memory");
}

// ---------------------------------------------------------------------------
// Prep: f32 -> f16 weight copies + fused biases
// ---------------------------------------------------------------------------
__global__ void lstm_prep_kernel(const float* __restrict__ Wx1, const float* __restrict__ bx1,
                                 const float* __restrict__ Wh1, const float* __restrict__ bh1,
                                 const float* __restrict__ Wx2, const float* __restrict__ bx2,
                                 const float* __restrict__ Wh2, const float* __restrict__ bh2,
                                 _Float16* wx1h, _Float16* wh1h, _Float16* wx2h, _Float16* wh2h,
                                 float* bias1, float* bias2) {
    int i = blockIdx.x * blockDim.x + threadIdx.x;
    if (i < GATES * IN_SZ) wx1h[i] = (_Float16)Wx1[i];
    if (i < GATES * HID) {
        wh1h[i] = (_Float16)Wh1[i];
        wx2h[i] = (_Float16)Wx2[i];
        wh2h[i] = (_Float16)Wh2[i];
    }
    if (i < GATES) {
        bias1[i] = bx1[i] + bh1[i];
        bias2[i] = bx2[i] + bh2[i];
    }
}

// ---------------------------------------------------------------------------
// Persistent fused 2-layer LSTM: one workgroup per 16-row batch tile.
// ---------------------------------------------------------------------------
__global__ void __launch_bounds__(TPB, 1)
lstm_persistent_kernel(const float* __restrict__ x,
                       const _Float16* __restrict__ wx1h,
                       const _Float16* __restrict__ wh1h,
                       const _Float16* __restrict__ wx2h,
                       const _Float16* __restrict__ wh2h,
                       const float* __restrict__ bias1,
                       const float* __restrict__ bias2,
                       const float* __restrict__ Wc,
                       const float* __restrict__ bc,
                       float* __restrict__ out) {
    extern __shared__ char smem[];
    _Float16* sWh1 = (_Float16*)(smem + LDS_WH1);
    _Float16* sWh2 = (_Float16*)(smem + LDS_WH2);
    float*    sG   = (float*)   (smem + LDS_GATES);
    _Float16* sH1  = (_Float16*)(smem + LDS_H1);
    _Float16* sH2  = (_Float16*)(smem + LDS_H2);
    _Float16* sX0  = (_Float16*)(smem + LDS_X0);   // two buffers: sX0 + p*XBUFH

    const int tid    = threadIdx.x;
    const int lane   = tid & 31;
    const int wv     = tid >> 5;          // wave id 0..15
    const int batch0 = blockIdx.x * BT;

    // ---- Stage recurrent weights to padded LDS via async DMA (no VGPR round-trip) ----
    for (int i = tid; i < GATES * (HID / 8); i += TPB) {    // 8192 x 16B chunks per matrix
        int row = i >> 4, c = i & 15;
        async_g2l_b128((uint32_t)(uintptr_t)(sWh1 + row * WROW + c * 8),
                       (uint64_t)(uintptr_t)(wh1h + i * 8));
        async_g2l_b128((uint32_t)(uintptr_t)(sWh2 + row * WROW + c * 8),
                       (uint64_t)(uintptr_t)(wh2h + i * 8));
    }
    // Zero h state (padded region included).
    for (int i = tid; i < BT * HROW; i += TPB) {
        sH1[i] = (_Float16)0.0f;
        sH2[i] = (_Float16)0.0f;
    }

    const int n0 = wv * 32;               // each wave owns 32 gate columns = 2 N-tiles

    // Per-wave register-resident B fragments of Wx1 (K=64) and Wx2 (K=128).
    v16h bwx1[2][2];
    v16h bwx2[2][4];
#pragma unroll
    for (int nt = 0; nt < 2; ++nt) {
#pragma unroll
        for (int kc = 0; kc < 2; ++kc)
            bwx1[nt][kc] = load_bfrag(wx1h, IN_SZ, n0 + nt * 16, kc * 32, lane);
#pragma unroll
        for (int kc = 0; kc < 4; ++kc)
            bwx2[nt][kc] = load_bfrag(wx2h, HID, n0 + nt * 16, kc * 32, lane);
    }

    float b1[2], b2[2];
#pragma unroll
    for (int nt = 0; nt < 2; ++nt) {
        b1[nt] = bias1[n0 + nt * 16 + (lane & 15)];
        b2[nt] = bias2[n0 + nt * 16 + (lane & 15)];
    }

    // Cell state in registers: thread owns elements idx = tid + j*TPB.
    float c1[4] = {0.f, 0.f, 0.f, 0.f};
    float c2[4] = {0.f, 0.f, 0.f, 0.f};

    // ---- x double-buffering: thread owns 2 consecutive x elements of row `wv` ----
    const size_t xbase = ((size_t)(batch0 + wv) * SEQ) * IN_SZ + (size_t)(lane * 2);
    const int xl = wv * XROW + lane * 2;  // LDS slot within a buffer (halves)

    float2 xr = *(const float2*)(x + xbase);              // x_0
    {   // stage x_0 into buffer 0, preload x_1
        v2h h2; h2[0] = (_Float16)xr.x; h2[1] = (_Float16)xr.y;
        *(v2h*)(sX0 + xl) = h2;
        xr = *(const float2*)(x + xbase + 1 * IN_SZ);     // x_1
    }

    wait_async0();          // weight DMA done (ASYNCcnt, not covered by barrier's DScnt)
    __syncthreads();

#pragma unroll 1
    for (int t = 0; t < SEQ; ++t) {
        const _Float16* sXcur = sX0 + (t & 1) * XBUFH;

        // ---- gates1 = x_t @ Wx1^T + h1 @ Wh1^T + (bx1+bh1) ----
#pragma unroll
        for (int nt = 0; nt < 2; ++nt) {
            v8f acc;
#pragma unroll
            for (int v = 0; v < 8; ++v) acc[v] = b1[nt];
#pragma unroll
            for (int kc = 0; kc < 2; ++kc)
                acc = wmma_f16(load_afrag(sXcur, XROW, kc * 32, lane), bwx1[nt][kc], acc);
#pragma unroll
            for (int kc = 0; kc < 4; ++kc)
                acc = wmma_f16(load_afrag(sH1, HROW, kc * 32, lane),
                               load_bfrag(sWh1, WROW, n0 + nt * 16, kc * 32, lane), acc);
            int n = n0 + nt * 16 + (lane & 15);
            int mb = (lane & 16) ? 8 : 0;
#pragma unroll
            for (int v = 0; v < 8; ++v) sG[(mb + v) * GATES + n] = acc[v];
        }
        __syncthreads();

        // ---- cell 1 + stage x_{t+1} into the other buffer ----
        {
            v2h h2; h2[0] = (_Float16)xr.x; h2[1] = (_Float16)xr.y;
            *(v2h*)(sX0 + ((t + 1) & 1) * XBUFH + xl) = h2;
        }
#pragma unroll
        for (int j = 0; j < 4; ++j) {
            int idx = tid + j * TPB;              // 0..2047: m = idx>>7, col = idx&127
            int m = idx >> 7, col = idx & 127;
            float gi = sG[m * GATES + col];
            float gf = sG[m * GATES + col + HID];
            float gg = sG[m * GATES + col + 2 * HID];
            float go = sG[m * GATES + col + 3 * HID];
            float c  = fsig(gf) * c1[j] + fsig(gi) * ftanh(gg);
            c1[j] = c;
            sH1[m * HROW + col] = (_Float16)(fsig(go) * ftanh(c));
        }
        __syncthreads();

        // ---- gates2 = h1 @ Wx2^T + h2 @ Wh2^T + (bx2+bh2) ----
#pragma unroll
        for (int nt = 0; nt < 2; ++nt) {
            v8f acc;
#pragma unroll
            for (int v = 0; v < 8; ++v) acc[v] = b2[nt];
#pragma unroll
            for (int kc = 0; kc < 4; ++kc)
                acc = wmma_f16(load_afrag(sH1, HROW, kc * 32, lane), bwx2[nt][kc], acc);
#pragma unroll
            for (int kc = 0; kc < 4; ++kc)
                acc = wmma_f16(load_afrag(sH2, HROW, kc * 32, lane),
                               load_bfrag(sWh2, WROW, n0 + nt * 16, kc * 32, lane), acc);
            int n = n0 + nt * 16 + (lane & 15);
            int mb = (lane & 16) ? 8 : 0;
#pragma unroll
            for (int v = 0; v < 8; ++v) sG[(mb + v) * GATES + n] = acc[v];
        }
        __syncthreads();

        // ---- cell 2 + preload x_{t+2} into registers ----
        {
            int tt = (t + 2 < SEQ) ? (t + 2) : (SEQ - 1);
            xr = *(const float2*)(x + xbase + (size_t)tt * IN_SZ);
            __builtin_prefetch(x + xbase + (size_t)((t + 4 < SEQ) ? (t + 4) : (SEQ - 1)) * IN_SZ, 0, 3);
        }
#pragma unroll
        for (int j = 0; j < 4; ++j) {
            int idx = tid + j * TPB;
            int m = idx >> 7, col = idx & 127;
            float gi = sG[m * GATES + col];
            float gf = sG[m * GATES + col + HID];
            float gg = sG[m * GATES + col + 2 * HID];
            float go = sG[m * GATES + col + 3 * HID];
            float c  = fsig(gf) * c2[j] + fsig(gi) * ftanh(gg);
            c2[j] = c;
            sH2[m * HROW + col] = (_Float16)(fsig(go) * ftanh(c));
        }
        __syncthreads();
    }

    // ---- classifier: logits = h2 @ Wc^T + bc  (16 rows x 4 classes) ----
    if (tid < BT * NCLS) {
        int r = tid & 15, cls = tid >> 4;
        float s = bc[cls];
#pragma unroll 4
        for (int k = 0; k < HID; ++k)
            s += (float)sH2[r * HROW + k] * Wc[cls * HID + k];
        out[(size_t)(batch0 + r) * NCLS + cls] = s;
    }
}

// ---------------------------------------------------------------------------
extern "C" void kernel_launch(void* const* d_in, const int* in_sizes, int n_in,
                              void* d_out, int out_size, void* d_ws, size_t ws_size,
                              hipStream_t stream) {
    const float* x   = (const float*)d_in[0];
    const float* Wx1 = (const float*)d_in[1];
    const float* bx1 = (const float*)d_in[2];
    const float* Wh1 = (const float*)d_in[3];
    const float* bh1 = (const float*)d_in[4];
    const float* Wx2 = (const float*)d_in[5];
    const float* bx2 = (const float*)d_in[6];
    const float* Wh2 = (const float*)d_in[7];
    const float* bh2 = (const float*)d_in[8];
    const float* Wc  = (const float*)d_in[9];
    const float* bc  = (const float*)d_in[10];
    float* out = (float*)d_out;

    char* ws = (char*)d_ws;
    _Float16* wx1h = (_Float16*)(ws + WS_WX1H);
    _Float16* wh1h = (_Float16*)(ws + WS_WH1H);
    _Float16* wx2h = (_Float16*)(ws + WS_WX2H);
    _Float16* wh2h = (_Float16*)(ws + WS_WH2H);
    float* bias1   = (float*)(ws + WS_BIAS1);
    float* bias2   = (float*)(ws + WS_BIAS2);

    lstm_prep_kernel<<<(GATES * HID + 255) / 256, 256, 0, stream>>>(
        Wx1, bx1, Wh1, bh1, Wx2, bx2, Wh2, bh2,
        wx1h, wh1h, wx2h, wh2h, bias1, bias2);

    lstm_persistent_kernel<<<BATCH / BT, TPB, LDS_TOTAL, stream>>>(
        x, wx1h, wh1h, wx2h, wh2h, bias1, bias2, Wc, bc, out);
}
